// DualTernaryLinear_62989990363233
// MI455X (gfx1250) — matrix operand order repaired and checked by reading, not compiled, not run on hardware
//
#include <hip/hip_runtime.h>
#include <stdint.h>

// ---------------------------------------------------------------------------
// DualTernaryLinear on MI455X (gfx1250, wave32, WMMA + async LDS DMA)
//
// y[m,n] = a1[n]*sum_k x[m,k]*t1[n,k] + a2[n]*sum_k x[m,k]*t2[n,k] + b[n]
//
// Stage 1: x (f32) -> f16, t1/t2 (i32 ternary, exact in f16) -> f16 workspace.
//          f16 working set = 134 MB -> resident in the 192 MB L2.
// Stage 2: 128x128 workgroup tile, 8 waves of 64x32 each (16 WMMA : 16 ds_b128
//          per k-step per wave).  Global->LDS staging via
//          GLOBAL_LOAD_ASYNC_TO_LDS_B128 with double-buffered LDS and
//          s_wait_asynccnt pipelining; f32 epilogue applies a1/a2/bias.
// ---------------------------------------------------------------------------

typedef __attribute__((ext_vector_type(16))) _Float16 v16h;
typedef __attribute__((ext_vector_type(8)))  _Float16 v8h;
typedef __attribute__((ext_vector_type(4)))  _Float16 v4h;
typedef __attribute__((ext_vector_type(8)))  float    v8f;
typedef __attribute__((ext_vector_type(4)))  float    v4f;
typedef __attribute__((ext_vector_type(4)))  int      v4i;

typedef __attribute__((address_space(3))) _Float16 lds_t;

union FragAB { v16h v; v8h h[2]; };

constexpr int K_DIM = 4096;          // D_IN
constexpr int N_DIM = 4096;          // D_OUT
constexpr int MT    = 128;           // M tile per workgroup
constexpr int NT    = 128;           // N tile per workgroup
constexpr int KT    = 32;            // K step
constexpr int LDP   = KT + 8;        // padded LDS row stride (80 B) -> conflict free
constexpr int TILE_H = 128 * LDP;    // halves per staged tile (A, B1 or B2)
constexpr int BUF_H  = 3 * TILE_H;   // halves per double-buffer slot

// async DMA: 16 B per lane, IOFFSET applies to both LDS and global addresses
#define ASYNC_B128(ldsp, gp)                                                   \
    asm volatile("global_load_async_to_lds_b128 %0, %1, off"                   \
                 :: "v"(ldsp), "v"(gp) : "memory")
#define ASYNC_B128_O16(ldsp, gp)                                               \
    asm volatile("global_load_async_to_lds_b128 %0, %1, off offset:16"         \
                 :: "v"(ldsp), "v"(gp) : "memory")

// ---------------- stage 1: conversions ----------------

__global__ void __launch_bounds__(256)
cvt_f32_to_f16_kernel(const float* __restrict__ s, _Float16* __restrict__ d, int n4) {
    int i = blockIdx.x * 256 + threadIdx.x;
    if (i < n4) {
        v4f f = ((const v4f*)s)[i];
        v4h h;
        h[0] = (_Float16)f[0]; h[1] = (_Float16)f[1];
        h[2] = (_Float16)f[2]; h[3] = (_Float16)f[3];
        ((v4h*)d)[i] = h;
    }
}

__global__ void __launch_bounds__(256)
cvt_i32_to_f16_kernel(const int* __restrict__ s, _Float16* __restrict__ d, int n4) {
    int i = blockIdx.x * 256 + threadIdx.x;
    if (i < n4) {
        v4i t = ((const v4i*)s)[i];
        v4h h;  // ternary {-1,0,1}: exact in f16
        h[0] = (_Float16)(float)t[0]; h[1] = (_Float16)(float)t[1];
        h[2] = (_Float16)(float)t[2]; h[3] = (_Float16)(float)t[3];
        ((v4h*)d)[i] = h;
    }
}

// ---------------- stage 2: dual-plane WMMA GEMM ----------------

__global__ void __launch_bounds__(256)
dual_ternary_wmma_kernel(const _Float16* __restrict__ Xh,   // [M, K]
                         const _Float16* __restrict__ W1h,  // [N, K]
                         const _Float16* __restrict__ W2h,  // [N, K]
                         const float* __restrict__ a1,
                         const float* __restrict__ a2,
                         const float* __restrict__ bias,
                         float* __restrict__ Y)              // [M, N]
{
    __shared__ _Float16 lds[2 * BUF_H];   // 60 KB, double buffered

    const int tid   = threadIdx.x;
    const int lane  = tid & 31;
    const int wave  = tid >> 5;       // 8 waves
    const int waveM = wave & 1;       // 2 waves along M (64 rows each)
    const int waveN = wave >> 1;      // 4 waves along N (32 cols each)
    const int m0    = blockIdx.y * MT;
    const int n0    = blockIdx.x * NT;
    const int halfw = lane >> 4;      // 0/1
    const int l15   = lane & 15;

    // staging map: thread owns one 32 B (16-half) segment: row sr, k-seg sk
    const int sr = tid >> 1;          // 0..127
    const int sk = (tid & 1) * 16;    // halves
    lds_t* ldsA  = ((lds_t*)lds) + sr * LDP + sk;
    lds_t* ldsB1 = ldsA + TILE_H;
    lds_t* ldsB2 = ldsA + 2 * TILE_H;
    const _Float16* xg  = Xh  + (size_t)(m0 + sr) * K_DIM + sk;
    const _Float16* w1g = W1h + (size_t)(n0 + sr) * K_DIM + sk;
    const _Float16* w2g = W2h + (size_t)(n0 + sr) * K_DIM + sk;

    // prologue: buffer 0 <- k=0  (6 async b128 per thread)
    ASYNC_B128(ldsA,  xg);  ASYNC_B128_O16(ldsA,  xg);
    ASYNC_B128(ldsB1, w1g); ASYNC_B128_O16(ldsB1, w1g);
    ASYNC_B128(ldsB2, w2g); ASYNC_B128_O16(ldsB2, w2g);

    v8f c1[4][2] = {};
    v8f c2[4][2] = {};

    constexpr int NIT = K_DIM / KT;   // 128
    int curH = 0;
    for (int it = 0; it < NIT; ++it) {
        const int nxtH = curH ^ BUF_H;
        // last iteration refills k=0 (dummy, never read) so the in-flight
        // count stays uniform at 6 and no address goes out of bounds
        const int kNext = (it + 1 < NIT) ? (it + 1) * KT : 0;

        ASYNC_B128(ldsA  + nxtH, xg  + kNext); ASYNC_B128_O16(ldsA  + nxtH, xg  + kNext);
        ASYNC_B128(ldsB1 + nxtH, w1g + kNext); ASYNC_B128_O16(ldsB1 + nxtH, w1g + kNext);
        ASYNC_B128(ldsB2 + nxtH, w2g + kNext); ASYNC_B128_O16(ldsB2 + nxtH, w2g + kNext);

        // async loads complete in order: <=6 outstanding => current buffer done
        asm volatile("s_wait_asynccnt 0x6" ::: "memory");
        __syncthreads();

        const _Float16* Ab  = lds + curH;
        const _Float16* B1b = Ab + TILE_H;
        const _Float16* B2b = Ab + 2 * TILE_H;

        // A fragments: this code's contiguous k order (not ISA table order) — half-wave picks K base {0,16};
        // halves [0..15]=K[kb..kb+15] (contiguous, matches B)
        const int kb = halfw * 16;
        FragAB a[4];
#pragma unroll
        for (int mi = 0; mi < 4; ++mi) {
            const _Float16* r = Ab + (waveM * 64 + mi * 16 + l15) * LDP;
            a[mi].h[0] = *(const v8h*)(r + kb);
            a[mi].h[1] = *(const v8h*)(r + kb + 8);
        }

        // B fragments: same k order as A (not ISA table order) — col = lane%16, contiguous
        // K[halfw*16 .. +15]
        const int kob = halfw * 16;
#pragma unroll
        for (int ni = 0; ni < 2; ++ni) {
            const int col = waveN * 32 + ni * 16 + l15;
            FragAB b1, b2;
            const _Float16* r1 = B1b + col * LDP + kob;
            b1.h[0] = *(const v8h*)r1; b1.h[1] = *(const v8h*)(r1 + 8);
            const _Float16* r2 = B2b + col * LDP + kob;
            b2.h[0] = *(const v8h*)r2; b2.h[1] = *(const v8h*)(r2 + 8);
#pragma unroll
            for (int mi = 0; mi < 4; ++mi) {
                c1[mi][ni] = __builtin_amdgcn_wmma_f32_16x16x32_f16(
                    false, a[mi].v, false, b1.v, (short)0, c1[mi][ni], false, false);
                c2[mi][ni] = __builtin_amdgcn_wmma_f32_16x16x32_f16(
                    false, a[mi].v, false, b2.v, (short)0, c2[mi][ni], false, false);
            }
        }

        __syncthreads();   // all waves done reading curH before it is refilled
        curH = nxtH;
    }

    // epilogue: D layout — lanes 0-15: M=v, lanes 16-31: M=v+8; N = lane%16
#pragma unroll
    for (int ni = 0; ni < 2; ++ni) {
        const int colG = n0 + waveN * 32 + ni * 16 + l15;
        const float s1 = a1[colG];
        const float s2 = a2[colG];
        const float bb = bias[colG];
#pragma unroll
        for (int mi = 0; mi < 4; ++mi) {
            const int rowBase = m0 + waveM * 64 + mi * 16 + halfw * 8;
#pragma unroll
            for (int v = 0; v < 8; ++v) {
                const size_t idx = (size_t)(rowBase + v) * N_DIM + colG;
                Y[idx] = c1[mi][ni][v] * s1 + c2[mi][ni][v] * s2 + bb;
            }
        }
    }
}

// ---------------- host launcher ----------------

extern "C" void kernel_launch(void* const* d_in, const int* in_sizes, int n_in,
                              void* d_out, int out_size, void* d_ws, size_t ws_size,
                              hipStream_t stream) {
    (void)n_in; (void)out_size; (void)ws_size;

    const float* x    = (const float*)d_in[0];
    const int*   t1   = (const int*)  d_in[1];
    const int*   t2   = (const int*)  d_in[2];
    const float* a1   = (const float*)d_in[3];
    const float* a2   = (const float*)d_in[4];
    const float* bias = (const float*)d_in[5];
    float*       y    = (float*)d_out;

    const int Mtot = in_sizes[0] / K_DIM;   // B*S = 8192

    // workspace: xh [M*K] f16, w1h [N*K] f16, w2h [N*K] f16 (~128 MiB)
    _Float16* xh  = (_Float16*)d_ws;
    _Float16* w1h = xh  + (size_t)Mtot  * K_DIM;
    _Float16* w2h = w1h + (size_t)N_DIM * K_DIM;

    const int nx4 = (Mtot  * K_DIM) / 4;
    const int nw4 = (N_DIM * K_DIM) / 4;
    cvt_f32_to_f16_kernel<<<(nx4 + 255) / 256, 256, 0, stream>>>(x,  xh,  nx4);
    cvt_i32_to_f16_kernel<<<(nw4 + 255) / 256, 256, 0, stream>>>(t1, w1h, nw4);
    cvt_i32_to_f16_kernel<<<(nw4 + 255) / 256, 256, 0, stream>>>(t2, w2h, nw4);

    dim3 grid(N_DIM / NT, Mtot / MT);   // 32 x 64 workgroups
    dual_ternary_wmma_kernel<<<grid, 256, 0, stream>>>(xh, w1h, w2h, a1, a2, bias, y);
}
